// MultiHeadAttention_62053687492911
// MI455X (gfx1250) — matrix operand order repaired and checked by reading, not compiled, not run on hardware
//
#include <hip/hip_runtime.h>
#include <hip/hip_bf16.h>

typedef __attribute__((ext_vector_type(16))) _Float16 v16h;
typedef __attribute__((ext_vector_type(8)))  _Float16 v8h;
typedef __attribute__((ext_vector_type(8)))  float    v8f;
typedef __attribute__((ext_vector_type(4)))  unsigned int u32x4;
typedef __attribute__((ext_vector_type(8)))  int    i32x8;
typedef __attribute__((ext_vector_type(4)))  int    i32x4;

#define B_  4
#define L_  2048
#define D_  512
#define H_  8
#define DH_ 64
#define M_  (B_ * L_)          // 8192 rows
#define NEG_BIG (-1.0e30f)

// ---------------- WMMA helpers (layouts per CDNA5 ISA 7.12.2) ----------------

__device__ __forceinline__ v8f wmma16(v16h a, v16h b, v8f c) {
  return __builtin_amdgcn_wmma_f32_16x16x32_f16(false, a, false, b, (short)0, c,
                                                false, false);
}

// A fragment: 16x32 tile, A row-major with leading dim lda (halves).
__device__ __forceinline__ v16h load_a_frag(const _Float16* A, int lda, int lane) {
  int m = lane & 15, hi = lane >> 4;
  const _Float16* row = A + (size_t)m * lda;
  v8h c0 = *(const v8h*)(row + hi * 8);
  v8h c1 = *(const v8h*)(row + 16 + hi * 8);
  v16h f;
  #pragma unroll
  for (int i = 0; i < 8; ++i) { f[i] = c0[i]; f[i + 8] = c1[i]; }
  return f;
}

// B fragment: 32x16 tile (KxN); value(k,n) = Bm[n*ldb + k].
__device__ __forceinline__ v16h load_b_frag(const _Float16* Bm, int ldb, int lane) {
  int n = lane & 15, hi = lane >> 4;
  const _Float16* p = Bm + (size_t)n * ldb;
  v8h c0 = *(const v8h*)(p + hi * 8);
  v8h c1 = *(const v8h*)(p + 16 + hi * 8);
  v16h f;
  #pragma unroll
  for (int i = 0; i < 8; ++i) { f[i] = c0[i]; f[i + 8] = c1[i]; }
  return f;
}

// ---------------- TDM: DMA a 64(rows)x32(k) f16 tile into LDS ----------------
// D# per CDNA5 ISA 8.3/8.4. 2D tile, data_size=2B, row stride = D_ elements.

__device__ __forceinline__ void tdm_load_tile64x32(const _Float16* gptr,
                                                   unsigned lds_off,
                                                   unsigned tensor_rows) {
  unsigned long long ga = (unsigned long long)(uintptr_t)gptr;
  u32x4 g0;
  g0[0] = 1u;                                                  // count=1
  g0[1] = lds_off;                                             // LDS byte addr
  g0[2] = (unsigned)(ga & 0xffffffffu);                        // global_addr lo
  g0[3] = (unsigned)((ga >> 32) & 0x01ffffffu) | 0x80000000u;  // addr hi | type=2
  i32x8 g1;
  g1[0] = 0x00010000;                              // data_size=1 (2 bytes)
  g1[1] = (int)(((unsigned)D_ & 0xffffu) << 16);   // tensor_dim0 = 512
  g1[2] = (int)((tensor_rows & 0xffffu) << 16);    // tensor_dim1 lo16
  g1[3] = (int)((tensor_rows >> 16) | (32u << 16));// dim1 hi16 | tile_dim0=32
  g1[4] = 64;                                      // tile_dim1=64, tile_dim2=0
  g1[5] = D_;                                      // tensor_dim0_stride = 512
  g1[6] = 0;
  g1[7] = 0;
  i32x4 z4 = {};
#if defined(__clang_major__) && (__clang_major__ >= 23)
  i32x8 z8 = {};
  __builtin_amdgcn_tensor_load_to_lds(g0, g1, z4, z4, z8, 0);
#else
  __builtin_amdgcn_tensor_load_to_lds(g0, g1, z4, z4, 0);
#endif
}

__device__ __forceinline__ unsigned lds_off_of(const void* p) {
  return (unsigned)(uintptr_t)p;   // LDS aperture: low 32 bits = byte offset
}

// ---------------- Kernel 1: LayerNorm -> f16 (wave per row) ----------------

__global__ __launch_bounds__(128) void mha_ln_kernel(const float* __restrict__ x,
                                                     const float* __restrict__ gamma,
                                                     const float* __restrict__ beta,
                                                     _Float16* __restrict__ xn) {
  int lane = threadIdx.x & 31;
  size_t row = (size_t)blockIdx.x * 4 + (threadIdx.x >> 5);
  const float* xr = x + row * D_;
  float4 v[4];
  float s = 0.f, ss = 0.f;
  #pragma unroll
  for (int i = 0; i < 4; ++i) {
    v[i] = ((const float4*)xr)[lane + i * 32];
    s  += v[i].x + v[i].y + v[i].z + v[i].w;
    ss += v[i].x * v[i].x + v[i].y * v[i].y + v[i].z * v[i].z + v[i].w * v[i].w;
  }
  #pragma unroll
  for (int off = 1; off < 32; off <<= 1) {
    s  += __shfl_xor(s, off, 32);
    ss += __shfl_xor(ss, off, 32);
  }
  float mu  = s * (1.f / D_);
  float var = ss * (1.f / D_) - mu * mu;
  float rst = rsqrtf(var + 1e-5f);
  #pragma unroll
  for (int i = 0; i < 4; ++i) {
    int j = (lane + i * 32) * 4;
    float e[4] = {v[i].x, v[i].y, v[i].z, v[i].w};
    #pragma unroll
    for (int c = 0; c < 4; ++c)
      xn[row * D_ + j + c] = (_Float16)((e[c] - mu) * rst * gamma[j + c] + beta[j + c]);
  }
}

// ---------------- Kernel 2: weight f32 -> f16 ----------------

__global__ void mha_wconv_kernel(const float* __restrict__ W,
                                 _Float16* __restrict__ W16, int n) {
  int i = blockIdx.x * 256 + threadIdx.x;
  if (i < n) W16[i] = (_Float16)W[i];
}

// ---------------- TDM-staged block GEMM core ----------------
// Block = 4 waves, 64x64 output tile. Wave w computes rows 16w..16w+15.
// Double-buffered LDS tiles filled by wave 0 via tensor_load_to_lds.

__device__ __forceinline__ void gemm_block_core(const _Float16* __restrict__ Abase,
                                                const _Float16* __restrict__ Wbase,
                                                _Float16 (*sA)[64 * 32],
                                                _Float16 (*sB)[64 * 32],
                                                int lane, int wid, v8f acc[4]) {
  if (wid == 0) {
    tdm_load_tile64x32(Abase, lds_off_of(&sA[0][0]), M_);
    tdm_load_tile64x32(Wbase, lds_off_of(&sB[0][0]), D_);
  }
  const int NCH = D_ / 32;   // 16 k-chunks
  for (int j = 0; j < NCH; ++j) {
    int p = j & 1;
    if (wid == 0) {
      if (j + 1 < NCH) {
        tdm_load_tile64x32(Abase + (j + 1) * 32, lds_off_of(&sA[p ^ 1][0]), M_);
        tdm_load_tile64x32(Wbase + (j + 1) * 32, lds_off_of(&sB[p ^ 1][0]), D_);
        __builtin_amdgcn_s_wait_tensorcnt(2);   // chunk j landed (in-order TDM)
      } else {
        __builtin_amdgcn_s_wait_tensorcnt(0);
      }
    }
    __syncthreads();
    v16h af = load_a_frag(&sA[p][wid * 16 * 32], 32, lane);
    #pragma unroll
    for (int t = 0; t < 4; ++t) {
      v16h bf = load_b_frag(&sB[p][t * 16 * 32], 32, lane);
      acc[t] = wmma16(af, bf, acc[t]);
    }
    __syncthreads();   // all waves done with buf p before it is refilled
  }
}

// ---------------- Kernel 3: QKV projection GEMM ----------------
// y[m,e] = sum_d xn[m,d] * W[e,d].
// vt_mode==0 -> store [B,H,L,Dh] ; vt_mode==1 -> store [B,H,Dh,L] (for V).

__global__ __launch_bounds__(128) void mha_qkv_gemm_kernel(const _Float16* __restrict__ A,
                                                           const _Float16* __restrict__ W,
                                                           _Float16* __restrict__ Out,
                                                           int vt_mode) {
  __shared__ _Float16 sA[2][64 * 32];
  __shared__ _Float16 sB[2][64 * 32];
  int lane = threadIdx.x & 31;
  int wid  = threadIdx.x >> 5;
  int tm = blockIdx.x >> 3;            // 128 row blocks of 64
  int tn = (blockIdx.x & 7) * 64;      // 8 column strips of 64
  v8f acc[4];
  #pragma unroll
  for (int t = 0; t < 4; ++t)
    #pragma unroll
    for (int i = 0; i < 8; ++i) acc[t][i] = 0.f;

  gemm_block_core(A + (size_t)tm * 64 * D_, W + (size_t)tn * D_, sA, sB, lane, wid, acc);

  int n = lane & 15, hi = lane >> 4;
  int row0 = tm * 64 + wid * 16;
  #pragma unroll
  for (int t = 0; t < 4; ++t) {
    int ng = tn + t * 16 + n;
    int hh = ng >> 6, dh = ng & 63;
    #pragma unroll
    for (int r = 0; r < 8; ++r) {
      int m = row0 + r + hi * 8;          // m = b*L + l
      int bb = m >> 11, l = m & (L_ - 1);
      size_t idx = vt_mode
                 ? ((size_t)((bb * H_ + hh) * DH_ + dh)) * L_ + l
                 : ((size_t)((bb * H_ + hh) * L_ + l)) * DH_ + dh;
      Out[idx] = (_Float16)acc[t][r];
    }
  }
}

// ---------------- Kernel 4: flash attention (wave per 16-query tile) ----------------

__global__ __launch_bounds__(128) void mha_attn_kernel(const _Float16* __restrict__ Q16,
                                                       const _Float16* __restrict__ K16,
                                                       const _Float16* __restrict__ Vt16,
                                                       const int* __restrict__ tok,
                                                       _Float16* __restrict__ O16) {
  __shared__ _Float16 pbuf[4][16 * 32];     // per-wave P tile bounce
  int lane = threadIdx.x & 31;
  int wid  = threadIdx.x >> 5;
  int tile = blockIdx.x * 4 + wid;          // 4096 tiles
  int qt = tile & 127;
  int h  = (tile >> 7) & 7;
  int b  = tile >> 10;
  int q0 = qt * 16;
  const _Float16* Qh = Q16  + ((size_t)(b * H_ + h)) * L_ * DH_;
  const _Float16* Kh = K16  + ((size_t)(b * H_ + h)) * L_ * DH_;
  const _Float16* Vh = Vt16 + ((size_t)(b * H_ + h)) * DH_ * L_;
  const int* tokb = tok + b * L_;

  int n = lane & 15, hi = lane >> 4;

  v16h qa0 = load_a_frag(Qh + (size_t)q0 * DH_,      DH_, lane);   // d 0..31
  v16h qa1 = load_a_frag(Qh + (size_t)q0 * DH_ + 32, DH_, lane);   // d 32..63

  v8f acc[4];
  float mrow[8], srow[8];
  #pragma unroll
  for (int t = 0; t < 4; ++t)
    #pragma unroll
    for (int i = 0; i < 8; ++i) acc[t][i] = 0.f;
  #pragma unroll
  for (int r = 0; r < 8; ++r) { mrow[r] = NEG_BIG; srow[r] = 0.f; }

  int nkb = (q0 + 16 + 31) >> 5;            // 32-key blocks covering keys <= q0+15
  for (int kb32 = 0; kb32 < nkb; ++kb32) {
    int kb = kb32 * 32;

    // prefetch next key block's K and V tiles (global_prefetch_b8)
    if (kb32 + 1 < nkb) {
      __builtin_prefetch(Kh + (size_t)(kb + 32) * DH_ + lane * DH_, 0, 1);
      __builtin_prefetch(Vh + (size_t)lane * L_ + kb + 32, 0, 1);
      __builtin_prefetch(Vh + (size_t)(lane + 32) * L_ + kb + 32, 0, 1);
    }

    // ---- scores: S[16 x 32] = Q * K^T / 8  (batch loads, then WMMAs) ----
    v16h kf[4];
    kf[0] = load_b_frag(Kh + (size_t)kb * DH_,             DH_, lane);
    kf[1] = load_b_frag(Kh + (size_t)kb * DH_ + 32,        DH_, lane);
    kf[2] = load_b_frag(Kh + (size_t)(kb + 16) * DH_,      DH_, lane);
    kf[3] = load_b_frag(Kh + (size_t)(kb + 16) * DH_ + 32, DH_, lane);
    v8f s0, s1;
    #pragma unroll
    for (int i = 0; i < 8; ++i) { s0[i] = 0.f; s1[i] = 0.f; }
    s0 = wmma16(qa0, kf[0], s0);
    s0 = wmma16(qa1, kf[1], s0);
    s1 = wmma16(qa0, kf[2], s1);
    s1 = wmma16(qa1, kf[3], s1);

    int key0 = kb + n, key1 = kb + 16 + n;
    bool pad0 = (tokb[key0] == 0);
    bool pad1 = (tokb[key1] == 0);

    // ---- online softmax per row (rows m = r + hi*8, cols across 16 lanes) ----
    #pragma unroll
    for (int r = 0; r < 8; ++r) {
      int qrow = q0 + r + hi * 8;
      float a = s0[r] * 0.125f; if (pad0 || key0 > qrow) a = NEG_BIG;
      float c = s1[r] * 0.125f; if (pad1 || key1 > qrow) c = NEG_BIG;
      float v = fmaxf(a, c);
      #pragma unroll
      for (int off = 1; off < 16; off <<= 1) v = fmaxf(v, __shfl_xor(v, off, 32));
      float mnew = fmaxf(mrow[r], v);
      float corr = __expf(mrow[r] - mnew);
      float p0 = __expf(a - mnew);
      float p1 = __expf(c - mnew);
      float ps = p0 + p1;
      #pragma unroll
      for (int off = 1; off < 16; off <<= 1) ps += __shfl_xor(ps, off, 32);
      srow[r] = srow[r] * corr + ps;
      mrow[r] = mnew;
      #pragma unroll
      for (int t = 0; t < 4; ++t) acc[t][r] *= corr;
      int m = r + hi * 8;
      pbuf[wid][m * 32 + n]      = (_Float16)p0;
      pbuf[wid][m * 32 + 16 + n] = (_Float16)p1;
    }
    // wave-private LDS bounce; DS ops from one wave are in-order
    asm volatile("s_wait_dscnt 0" ::: "memory");
    v16h pf = load_a_frag(&pbuf[wid][0], 32, lane);
    asm volatile("" ::: "memory");
    // ---- acc += P * V  (Vt layout: value(k=key, n=d) = Vt[d*L + key]) ----
    v16h vf[4];
    #pragma unroll
    for (int t = 0; t < 4; ++t)
      vf[t] = load_b_frag(Vh + (size_t)(t * 16) * L_ + kb, L_, lane);
    #pragma unroll
    for (int t = 0; t < 4; ++t) acc[t] = wmma16(pf, vf[t], acc[t]);
  }

  // ---- finalize + store attention output as f16 [B*L, D] ----
  #pragma unroll
  for (int r = 0; r < 8; ++r) {
    float inv = (srow[r] > 0.f) ? (1.f / srow[r]) : 0.f;
    int m = r + hi * 8;
    size_t base = ((size_t)(b * L_) + q0 + m) * D_ + h * DH_;
    #pragma unroll
    for (int t = 0; t < 4; ++t)
      O16[base + t * 16 + n] = (_Float16)(acc[t][r] * inv);
  }
}

// ---------------- Kernel 5: output projection + residual (f32 out) ----------------

__global__ __launch_bounds__(128) void mha_oproj_kernel(const _Float16* __restrict__ A,
                                                        const _Float16* __restrict__ Wo,
                                                        const float* __restrict__ X,
                                                        float* __restrict__ Out) {
  __shared__ _Float16 sA[2][64 * 32];
  __shared__ _Float16 sB[2][64 * 32];
  int lane = threadIdx.x & 31;
  int wid  = threadIdx.x >> 5;
  int tm = blockIdx.x >> 3;
  int tn = (blockIdx.x & 7) * 64;
  v8f acc[4];
  #pragma unroll
  for (int t = 0; t < 4; ++t)
    #pragma unroll
    for (int i = 0; i < 8; ++i) acc[t][i] = 0.f;

  gemm_block_core(A + (size_t)tm * 64 * D_, Wo + (size_t)tn * D_, sA, sB, lane, wid, acc);

  int n = lane & 15, hi = lane >> 4;
  int row0 = tm * 64 + wid * 16;
  #pragma unroll
  for (int t = 0; t < 4; ++t) {
    int ng = tn + t * 16 + n;
    #pragma unroll
    for (int r = 0; r < 8; ++r) {
      size_t m = (size_t)(row0 + r + hi * 8);
      Out[m * D_ + ng] = X[m * D_ + ng] + acc[t][r];
    }
  }
}

// ---------------- host launcher ----------------

extern "C" void kernel_launch(void* const* d_in, const int* in_sizes, int n_in,
                              void* d_out, int out_size, void* d_ws, size_t ws_size,
                              hipStream_t stream) {
  const float* x     = (const float*)d_in[0];
  const int*   tok   = (const int*)d_in[1];
  const float* Wq    = (const float*)d_in[2];
  const float* Wk    = (const float*)d_in[3];
  const float* Wv    = (const float*)d_in[4];
  const float* Wo    = (const float*)d_in[5];
  const float* gamma = (const float*)d_in[6];
  const float* beta  = (const float*)d_in[7];
  float* out = (float*)d_out;

  const size_t MD   = (size_t)M_ * D_;        // 8192*512
  const size_t WSZ  = (size_t)D_ * D_;        // 512*512
  const size_t BHLD = (size_t)B_ * H_ * L_ * DH_;

  _Float16* xn16 = (_Float16*)d_ws;           // MD halves; later reused as attn-out
  _Float16* w16  = xn16 + MD;                 // 4*WSZ halves (q,k,v,o)
  _Float16* q16  = w16 + 4 * WSZ;
  _Float16* k16  = q16 + BHLD;
  _Float16* vt16 = k16 + BHLD;

  // 1) LayerNorm -> f16
  mha_ln_kernel<<<M_ / 4, 128, 0, stream>>>(x, gamma, beta, xn16);

  // 2) weights -> f16
  const float* Ws[4] = {Wq, Wk, Wv, Wo};
  for (int j = 0; j < 4; ++j)
    mha_wconv_kernel<<<(int)(WSZ / 256), 256, 0, stream>>>(Ws[j], w16 + j * WSZ, (int)WSZ);

  // 3) Q, K (layout [B,H,L,Dh]) and V (transposed [B,H,Dh,L])
  mha_qkv_gemm_kernel<<<1024, 128, 0, stream>>>(xn16, w16 + 0 * WSZ, q16, 0);
  mha_qkv_gemm_kernel<<<1024, 128, 0, stream>>>(xn16, w16 + 1 * WSZ, k16, 0);
  mha_qkv_gemm_kernel<<<1024, 128, 0, stream>>>(xn16, w16 + 2 * WSZ, vt16, 1);

  // 4) flash attention -> f16 [B*L, D] (reuses xn16 buffer)
  mha_attn_kernel<<<1024, 128, 0, stream>>>(q16, k16, vt16, tok, xn16);

  // 5) output projection + residual -> f32
  mha_oproj_kernel<<<1024, 128, 0, stream>>>(xn16, w16 + 3 * WSZ, x, out);
}
